// FKD_71829033058971
// MI455X (gfx1250) — matrix operand order repaired and checked by reading, not compiled, not run on hardware
//
#include <hip/hip_runtime.h>
#include <hip/hip_bf16.h>

typedef __bf16 bf16;
typedef __attribute__((ext_vector_type(16))) __bf16 v16bf;
typedef __attribute__((ext_vector_type(8)))  __bf16 v8bf;
typedef __attribute__((ext_vector_type(8)))  float  v8f;

#define TILE_I 128
#define TILE_J 64
#define TILE_K 32
#define LSTR   40   // LDS row stride in bf16 elements (32 data + 8 pad) -> conflict-free

// Split a float into hi/lo bf16 and store a packed pair (2 elems) into both LDS tiles.
static __device__ __forceinline__ void split_store2(bf16* hi, bf16* lo, int idx,
                                                    float x0, float x1) {
    bf16 h0 = (bf16)x0;
    bf16 h1 = (bf16)x1;
    bf16 l0 = (bf16)(x0 - (float)h0);   // residual exact in f32, then RN to bf16
    bf16 l1 = (bf16)(x1 - (float)h1);
    union { bf16 b[2]; unsigned int u; } ph, pl;
    ph.b[0] = h0; ph.b[1] = h1;
    pl.b[0] = l0; pl.b[1] = l1;
    *(unsigned int*)(hi + idx) = ph.u;
    *(unsigned int*)(lo + idx) = pl.u;
}

// Assemble a v16bf fragment from two 8-element (16B) LDS reads.
static __device__ __forceinline__ v16bf frag16(const bf16* p, int off0, int off1) {
    v8bf a = *(const v8bf*)(p + off0);
    v8bf b = *(const v8bf*)(p + off1);
    return __builtin_shufflevector(a, b, 0,1,2,3,4,5,6,7,8,9,10,11,12,13,14,15);
}

// C = op(A[I,K] * B[J,K]^T):
//   MODE 0: C[i,j] = dot + cvec[j]                (projector GEMM, cvec = bias)
//   MODE 1: C[i,j] = rvec[i] + cvec[j] - 2*dot    (squared-distance GEMM)
// Requires I % 128 == 0, J % 64 == 0, K % 32 == 0; lda = ldb = K.
template<int MODE>
__global__ __launch_bounds__(256)
void gemm_bt_kernel(const float* __restrict__ A, const float* __restrict__ B,
                    float* __restrict__ C,
                    const float* __restrict__ rvec, const float* __restrict__ cvec,
                    int K, int ldc)
{
    __shared__ __align__(16) bf16 Ah[TILE_I * LSTR];
    __shared__ __align__(16) bf16 Al[TILE_I * LSTR];
    __shared__ __align__(16) bf16 Bh[TILE_J * LSTR];
    __shared__ __align__(16) bf16 Bl[TILE_J * LSTR];

    const int tid  = threadIdx.x;
    const int lane = tid & 31;
    const int wave = tid >> 5;
    const int wr = (wave >> 1) * 32;   // wave row base within tile: 0/32/64/96
    const int wc = (wave & 1) * 32;    // wave col base within tile: 0/32
    const int ln = lane & 15;
    const int hv = lane >> 4;          // half-wave select (0/1)

    const size_t iBase = (size_t)blockIdx.y * TILE_I;
    const size_t jBase = (size_t)blockIdx.x * TILE_J;
    const int nk = K / TILE_K;

    float4 aF[4], bF[2];

    // prologue: stage k-tile 0 into registers
    {
        const float* Ab = A + iBase * (size_t)K;
        const float* Bb = B + jBase * (size_t)K;
#pragma unroll
        for (int q = 0; q < 4; ++q) {
            int idx = tid + 256 * q; int r = idx >> 3; int c4 = idx & 7;
            aF[q] = *(const float4*)(Ab + (size_t)r * K + c4 * 4);
        }
#pragma unroll
        for (int q = 0; q < 2; ++q) {
            int idx = tid + 256 * q; int r = idx >> 3; int c4 = idx & 7;
            bF[q] = *(const float4*)(Bb + (size_t)r * K + c4 * 4);
        }
    }

    v8f acc[2][2];
#pragma unroll
    for (int rt = 0; rt < 2; ++rt)
#pragma unroll
        for (int ct = 0; ct < 2; ++ct) acc[rt][ct] = (v8f){};

    for (int kt = 0; kt < nk; ++kt) {
        __syncthreads();    // previous compute finished reading LDS
        // split f32 -> (hi,lo) bf16 and store to LDS
#pragma unroll
        for (int q = 0; q < 4; ++q) {
            int idx = tid + 256 * q; int r = idx >> 3; int cc = (idx & 7) * 4;
            int base = r * LSTR + cc;
            split_store2(Ah, Al, base,     aF[q].x, aF[q].y);
            split_store2(Ah, Al, base + 2, aF[q].z, aF[q].w);
        }
#pragma unroll
        for (int q = 0; q < 2; ++q) {
            int idx = tid + 256 * q; int r = idx >> 3; int cc = (idx & 7) * 4;
            int base = r * LSTR + cc;
            split_store2(Bh, Bl, base,     bF[q].x, bF[q].y);
            split_store2(Bh, Bl, base + 2, bF[q].z, bF[q].w);
        }
        __syncthreads();

        // overlap: issue next k-tile's global loads before the WMMA burst
        if (kt + 1 < nk) {
            int k0 = (kt + 1) * TILE_K;
            const float* Ab = A + iBase * (size_t)K + k0;
            const float* Bb = B + jBase * (size_t)K + k0;
#pragma unroll
            for (int q = 0; q < 4; ++q) {
                int idx = tid + 256 * q; int r = idx >> 3; int c4 = idx & 7;
                aF[q] = *(const float4*)(Ab + (size_t)r * K + c4 * 4);
            }
#pragma unroll
            for (int q = 0; q < 2; ++q) {
                int idx = tid + 256 * q; int r = idx >> 3; int c4 = idx & 7;
                bF[q] = *(const float4*)(Bb + (size_t)r * K + c4 * 4);
            }
        }

        // B fragments: lane<16 holds K0..15, lane>=16 holds K16..31 (contiguous k order, same as A)
        v16bf bh[2], bl[2];
#pragma unroll
        for (int ct = 0; ct < 2; ++ct) {
            int base = (wc + ct * 16 + ln) * LSTR + hv * 16;
            bh[ct] = frag16(Bh, base, base + 8);
            bl[ct] = frag16(Bl, base, base + 8);
        }
        // A fragments: lane<16 holds K0..15, lane>=16 holds K16..31
#pragma unroll
        for (int rt = 0; rt < 2; ++rt) {
            int base = (wr + rt * 16 + ln) * LSTR + hv * 16;
            v16bf ah = frag16(Ah, base, base + 8);
            v16bf al = frag16(Al, base, base + 8);
#pragma unroll
            for (int ct = 0; ct < 2; ++ct) {
                acc[rt][ct] = __builtin_amdgcn_wmma_f32_16x16x32_bf16(
                    false, ah, false, bh[ct], (short)0, acc[rt][ct], false, false);
                acc[rt][ct] = __builtin_amdgcn_wmma_f32_16x16x32_bf16(
                    false, ah, false, bl[ct], (short)0, acc[rt][ct], false, false);
                acc[rt][ct] = __builtin_amdgcn_wmma_f32_16x16x32_bf16(
                    false, al, false, bh[ct], (short)0, acc[rt][ct], false, false);
            }
        }
    }

    // Epilogue. C/D layout: VGPR v, lane<16 -> (M=v, N=lane); lane>=16 -> (M=v+8, N=lane-16)
#pragma unroll
    for (int rt = 0; rt < 2; ++rt) {
#pragma unroll
        for (int ct = 0; ct < 2; ++ct) {
            size_t col = jBase + wc + ct * 16 + ln;
            float cv = cvec[col];
            v8f c = acc[rt][ct];
#pragma unroll
            for (int v = 0; v < 8; ++v) {
                size_t row = iBase + wr + rt * 16 + v + 8 * hv;
                float val;
                if (MODE == 0) val = c[v] + cv;
                else           val = rvec[row] + cv - 2.0f * c[v];
                C[row * (size_t)ldc + col] = val;
            }
        }
    }
}

// out[row] = sum_k X[row][k]^2
__global__ __launch_bounds__(256)
void rowsq_kernel(const float* __restrict__ X, float* __restrict__ out, int K)
{
    const int row = blockIdx.x;
    const float4* x4 = (const float4*)(X + (size_t)row * K);
    float s = 0.f;
    for (int i = threadIdx.x; i < K / 4; i += 256) {
        float4 v = x4[i];
        s += v.x * v.x + v.y * v.y + v.z * v.z + v.w * v.w;
    }
    __shared__ float red[256];
    red[threadIdx.x] = s;
    __syncthreads();
#pragma unroll
    for (int w = 128; w >= 1; w >>= 1) {
        if (threadIdx.x < w) red[threadIdx.x] += red[threadIdx.x + w];
        __syncthreads();
    }
    if (threadIdx.x == 0) out[row] = red[0];
}

extern "C" void kernel_launch(void* const* d_in, const int* in_sizes, int n_in,
                              void* d_out, int out_size, void* d_ws, size_t ws_size,
                              hipStream_t stream)
{
    const float* t_rep = (const float*)d_in[0];   // [N, D]
    const float* s_rep = (const float*)d_in[1];   // [M, D]
    const float* W     = (const float*)d_in[2];   // [D, D]
    const float* bias  = (const float*)d_in[3];   // [D]
    float* out = (float*)d_out;                   // [N, M]

    const int D = in_sizes[3];          // 1536
    const int N = in_sizes[0] / D;      // 4096
    const int M = in_sizes[1] / D;      // 4096

    float* s_proj = (float*)d_ws;                 // [M, D]
    float* t_sq   = s_proj + (size_t)M * D;       // [N]
    float* s_sq   = t_sq + N;                     // [M]

    dim3 blk(256);

    // 1) s = s_rep @ W^T + bias    (I=M, J=D, K=D)
    gemm_bt_kernel<0><<<dim3(D / TILE_J, M / TILE_I), blk, 0, stream>>>(
        s_rep, W, s_proj, nullptr, bias, D, D);

    // 2) row squared norms
    rowsq_kernel<<<dim3(N), blk, 0, stream>>>(t_rep, t_sq, D);
    rowsq_kernel<<<dim3(M), blk, 0, stream>>>(s_proj, s_sq, D);

    // 3) out = t_sq[i] + s_sq[j] - 2 * (t_rep @ s^T)   (I=N, J=M, K=D)
    gemm_bt_kernel<1><<<dim3(M / TILE_J, N / TILE_I), blk, 0, stream>>>(
        t_rep, s_proj, out, t_sq, s_sq, D, M);
}
